// DeformableAttentionTransformerBlock_82781199663250
// MI455X (gfx1250) — matrix operand order repaired and checked by reading, not compiled, not run on hardware
//
#include <hip/hip_runtime.h>
#include <cstdint>
#include <cstddef>

// ---------------------------------------------------------------- constants
#define C_DIM   384
#define NSEQ    1024
#define BATCH   16
#define BN      (BATCH * NSEQ)      // 16384 rows
#define HEADS   12
#define HD      32
#define MLP_DIM 1536

typedef __attribute__((ext_vector_type(16))) __bf16 v16bf;
typedef __attribute__((ext_vector_type(8)))  __bf16 v8bf;
typedef __attribute__((ext_vector_type(8)))  float  v8f;

// ---------------------------------------------------------------- helpers
__device__ __forceinline__ float gelu_f(float x) {
  // tanh approximation (jax.nn.gelu default approximate=True)
  float x3 = x * x * x;
  return 0.5f * x * (1.0f + tanhf(0.7978845608028654f * (x + 0.044715f * x3)));
}

// A fragment: 16x32 bf16, row-major with leading dim ld.
// lane<16: row=lane, K = 0..7 and 16..23 ; lane>=16: row=lane-16, K = 8..15 and 24..31
__device__ __forceinline__ v16bf frag_a(const __bf16* base, int ld) {
  int lane = threadIdx.x & 31;
  const __bf16* p = base + (size_t)(lane & 15) * ld + ((lane >> 4) << 3);
  v8bf lo = *(const v8bf*)p;        // K = kb .. kb+7
  v8bf hi = *(const v8bf*)(p + 16); // K = kb+16 .. kb+23
  v16bf a;
#pragma unroll
  for (int i = 0; i < 8; ++i) { a[i] = lo[i]; a[i + 8] = hi[i]; }
  return a;
}

// B fragment: 32x16 bf16 (KxN) where column n's K-run is contiguous at base + n*ld.
// lane<16: col=lane, K = 0..7 and 16..23 ; lane>=16: col=lane-16, K = 8..15 and 24..31
__device__ __forceinline__ v16bf frag_b(const __bf16* base, int ld) {
  int lane = threadIdx.x & 31;
  const __bf16* p = base + (size_t)(lane & 15) * ld + ((lane >> 4) << 3);
  v8bf lo = *(const v8bf*)p;
  v8bf hi = *(const v8bf*)(p + 16);
  v16bf b;
#pragma unroll
  for (int i = 0; i < 8; ++i) { b[i] = lo[i]; b[i + 8] = hi[i]; }
  return b;
}

// block-wide sum over 128 threads (4 waves). scratch: >=4 floats shared.
__device__ __forceinline__ float block_sum(float v, float* scratch) {
#pragma unroll
  for (int m = 16; m >= 1; m >>= 1) v += __shfl_xor(v, m);
  int w = threadIdx.x >> 5;
  if ((threadIdx.x & 31) == 0) scratch[w] = v;
  __syncthreads();
  float r = 0.0f;
  int nw = blockDim.x >> 5;
  for (int i = 0; i < nw; ++i) r += scratch[i];
  __syncthreads();
  return r;
}

// ---------------------------------------------------------------- fp32 -> bf16
__global__ __launch_bounds__(256) void cvt_bf16_kernel(const float* __restrict__ s,
                                                       __bf16* __restrict__ d, int n) {
  int i = blockIdx.x * blockDim.x + threadIdx.x;
  if (i < n) d[i] = (__bf16)s[i];
}

// ---------------------------------------------------------------- LayerNorm
// one block (128 threads) per row of 384; optional f32 and bf16 outputs
__global__ __launch_bounds__(128) void ln_kernel(const float* __restrict__ x,
                                                 const float* __restrict__ g,
                                                 const float* __restrict__ bt,
                                                 float* __restrict__ outF,
                                                 __bf16* __restrict__ outB) {
  __shared__ float scratch[4];
  int row = blockIdx.x;
  int t = threadIdx.x;
  const float* xr = x + (size_t)row * C_DIM;
  float v[3], s = 0.0f, sq = 0.0f;
#pragma unroll
  for (int i = 0; i < 3; ++i) {
    v[i] = xr[t + 128 * i];
    s += v[i]; sq += v[i] * v[i];
  }
  s  = block_sum(s,  scratch);
  sq = block_sum(sq, scratch);
  float mean = s * (1.0f / C_DIM);
  float inv  = rsqrtf(sq * (1.0f / C_DIM) - mean * mean + 1e-5f);
#pragma unroll
  for (int i = 0; i < 3; ++i) {
    int c = t + 128 * i;
    float o = (v[i] - mean) * inv * g[c] + bt[c];
    if (outF) outF[(size_t)row * C_DIM + c] = o;
    if (outB) outB[(size_t)row * C_DIM + c] = (__bf16)o;
  }
}

// ---------------------------------------------------------------- WMMA GEMM
// D(MxN) = A(MxK,bf16) @ W^T + bias, weights W stored row-major (N x K) so that
// B[k][n] = W[n][k] gives contiguous-in-K per-lane loads.
// Each wave computes a 32x64 output patch (2x4 WMMA tiles), with software-
// pipelined (double-buffered) fragment loads so global loads overlap WMMA.
// EPI: 1 = f32 + bf16 out, 2 = bf16 out, 3 = bf16 out transposed (C-major per batch),
//      4 = gelu -> bf16, 5 = +resid -> f32
template <int EPI>
__global__ __launch_bounds__(256) void gemm_bf16_kernel(
    const __bf16* __restrict__ A, const __bf16* __restrict__ W,
    const float* __restrict__ bias, float* __restrict__ Df,
    __bf16* __restrict__ Db, const float* __restrict__ resid,
    int M, int N, int K) {
  const int wid   = (int)((blockIdx.x * blockDim.x + threadIdx.x) >> 5);
  const int lane  = threadIdx.x & 31;
  const int ncols = N >> 6;
  const int mrows = M >> 5;
  if (wid >= ncols * mrows) return;   // wave-uniform: EXEC stays all-ones
  const int mt = wid / ncols;
  const int nc = wid % ncols;

  v8f acc[2][4];
#pragma unroll
  for (int i = 0; i < 2; ++i)
#pragma unroll
    for (int j = 0; j < 4; ++j)
#pragma unroll
      for (int e = 0; e < 8; ++e) acc[i][j][e] = 0.0f;

  const __bf16* A0 = A + (size_t)(mt * 32) * K;
  const __bf16* A1 = A0 + (size_t)16 * K;
  const __bf16* W0 = W + (size_t)(nc * 64) * K;

  // prologue: load K-step 0 fragments
  v16bf ca0 = frag_a(A0, K);
  v16bf ca1 = frag_a(A1, K);
  v16bf cb[4];
#pragma unroll
  for (int j = 0; j < 4; ++j) cb[j] = frag_b(W0 + (size_t)(j * 16) * K, K);

  // steady state: issue next step's loads, then WMMA on current fragments
  for (int kt = 32; kt < K; kt += 32) {
    v16bf na0 = frag_a(A0 + kt, K);
    v16bf na1 = frag_a(A1 + kt, K);
    v16bf nb[4];
#pragma unroll
    for (int j = 0; j < 4; ++j) nb[j] = frag_b(W0 + (size_t)(j * 16) * K + kt, K);
#pragma unroll
    for (int j = 0; j < 4; ++j) {
      acc[0][j] = __builtin_amdgcn_wmma_f32_16x16x32_bf16(false, ca0, false, cb[j],
                                                          (short)0, acc[0][j], false, false);
      acc[1][j] = __builtin_amdgcn_wmma_f32_16x16x32_bf16(false, ca1, false, cb[j],
                                                          (short)0, acc[1][j], false, false);
    }
    ca0 = na0; ca1 = na1;
#pragma unroll
    for (int j = 0; j < 4; ++j) cb[j] = nb[j];
  }
  // epilogue K-step
#pragma unroll
  for (int j = 0; j < 4; ++j) {
    acc[0][j] = __builtin_amdgcn_wmma_f32_16x16x32_bf16(false, ca0, false, cb[j],
                                                        (short)0, acc[0][j], false, false);
    acc[1][j] = __builtin_amdgcn_wmma_f32_16x16x32_bf16(false, ca1, false, cb[j],
                                                        (short)0, acc[1][j], false, false);
  }

  const int col0 = lane & 15;
  const int roff = (lane >> 4) << 3;
#pragma unroll
  for (int i = 0; i < 2; ++i) {
#pragma unroll
    for (int j = 0; j < 4; ++j) {
      const int col = nc * 64 + j * 16 + col0;
      const float bb = bias[col];
#pragma unroll
      for (int r = 0; r < 8; ++r) {
        const int gm = mt * 32 + i * 16 + roff + r;
        float v = acc[i][j][r] + bb;
        if (EPI == 4) v = gelu_f(v);
        if (EPI == 5) v += resid[(size_t)gm * N + col];
        if (EPI == 1 || EPI == 5) Df[(size_t)gm * N + col] = v;
        if (EPI == 1 || EPI == 2 || EPI == 4) Db[(size_t)gm * N + col] = (__bf16)v;
        if (EPI == 3) {   // store transposed: vT[b][col][seq]
          const int bi = gm >> 10, seq = gm & 1023;
          Db[((size_t)bi * C_DIM + col) * NSEQ + seq] = (__bf16)v;
        }
      }
    }
  }
}

// ---------------------------------------------------------------- offset net + grid sample
// one block (128 threads) per pixel: depthwise 5x5 conv on q, LN, GELU, 1x1 -> 2,
// tanh * 0.125, pos = ref + off, bilinear sample xn -> xs (bf16)
__global__ __launch_bounds__(128) void offset_sample_kernel(
    const float* __restrict__ q, const float* __restrict__ off_k,
    const float* __restrict__ off_kb, const float* __restrict__ offg,
    const float* __restrict__ offb, const float* __restrict__ Woff,
    const float* __restrict__ boff, const float* __restrict__ xn,
    __bf16* __restrict__ xs) {
  __shared__ float scratch[4];
  __shared__ int   si[4];
  __shared__ float sf[2];
  const int pix = blockIdx.x;
  const int b = pix >> 10;
  const int p = pix & 1023;
  const int y = p >> 5;
  const int xx = p & 31;
  const int t = threadIdx.x;

  float cv[3];
#pragma unroll
  for (int i = 0; i < 3; ++i) {
    const int c = t + 128 * i;
    float acc = off_kb[c];
    for (int ky = 0; ky < 5; ++ky) {
      const int iy = y + ky - 2;
      if (iy < 0 || iy > 31) continue;
      for (int kx = 0; kx < 5; ++kx) {
        const int ix = xx + kx - 2;
        if (ix < 0 || ix > 31) continue;
        acc += q[((size_t)(b * 1024 + iy * 32 + ix)) * C_DIM + c] *
               off_k[(ky * 5 + kx) * C_DIM + c];
      }
    }
    cv[i] = acc;
  }
  float s = 0.0f, sq = 0.0f;
#pragma unroll
  for (int i = 0; i < 3; ++i) { s += cv[i]; sq += cv[i] * cv[i]; }
  s  = block_sum(s,  scratch);
  sq = block_sum(sq, scratch);
  const float mean = s * (1.0f / C_DIM);
  const float inv  = rsqrtf(sq * (1.0f / C_DIM) - mean * mean + 1e-5f);

  float d0 = 0.0f, d1 = 0.0f;
#pragma unroll
  for (int i = 0; i < 3; ++i) {
    const int c = t + 128 * i;
    const float n = (cv[i] - mean) * inv * offg[c] + offb[c];
    const float gl = gelu_f(n);
    d0 += gl * Woff[c];
    d1 += gl * Woff[C_DIM + c];
  }
  d0 = block_sum(d0, scratch);
  d1 = block_sum(d1, scratch);

  if (t == 0) {
    const float o0 = tanhf(d0 + boff[0]) * 0.125f;  // OFF_RANGE*2/H
    const float o1 = tanhf(d1 + boff[1]) * 0.125f;
    const float posy = (y + 0.5f) * (2.0f / 32.0f) - 1.0f + o0;
    const float posx = (xx + 0.5f) * (2.0f / 32.0f) - 1.0f + o1;
    const float py = (posy + 1.0f) * 0.5f * 31.0f;
    const float px = (posx + 1.0f) * 0.5f * 31.0f;
    const float fy = floorf(py), fx = floorf(px);
    int y0 = (int)fy; y0 = y0 < 0 ? 0 : (y0 > 31 ? 31 : y0);
    int x0 = (int)fx; x0 = x0 < 0 ? 0 : (x0 > 31 ? 31 : x0);
    si[0] = y0; si[1] = (y0 + 1 > 31) ? 31 : y0 + 1;
    si[2] = x0; si[3] = (x0 + 1 > 31) ? 31 : x0 + 1;
    sf[0] = py - fy; sf[1] = px - fx;
  }
  __syncthreads();
  const int y0 = si[0], y1 = si[1], x0 = si[2], x1 = si[3];
  const float wy = sf[0], wx = sf[1];
  const float w00 = (1.0f - wy) * (1.0f - wx), w01 = (1.0f - wy) * wx;
  const float w10 = wy * (1.0f - wx),          w11 = wy * wx;
  const size_t base = (size_t)(b * 1024) * C_DIM;
  const float* p00 = xn + base + (size_t)(y0 * 32 + x0) * C_DIM;
  const float* p01 = xn + base + (size_t)(y0 * 32 + x1) * C_DIM;
  const float* p10 = xn + base + (size_t)(y1 * 32 + x0) * C_DIM;
  const float* p11 = xn + base + (size_t)(y1 * 32 + x1) * C_DIM;
  __bf16* out = xs + (size_t)pix * C_DIM;
#pragma unroll
  for (int i = 0; i < 3; ++i) {
    const int c = t + 128 * i;
    out[c] = (__bf16)(p00[c] * w00 + p01[c] * w01 + p10[c] * w10 + p11[c] * w11);
  }
}

// ---------------------------------------------------------------- attention
// one block (4 waves) per (b, head, 16-query-row tile).
// S(16x1024) via WMMA -> LDS f32; softmax (unnormalized exp -> bf16 P in LDS);
// O(16x32) = P @ V via WMMA with K split over waves + LDS reduction.
#define ATTN_SMEM (16 * 1024 * 4 + 16 * 1024 * 2 + 1024 * 4 + 16 * 4)
__global__ __launch_bounds__(128) void attn_kernel(const __bf16* __restrict__ q,
                                                   const __bf16* __restrict__ k,
                                                   const __bf16* __restrict__ vT,
                                                   __bf16* __restrict__ o) {
  extern __shared__ char smem[];
  float*  sS   = (float*)smem;                                   // 16 x 1024 f32
  __bf16* sP   = (__bf16*)(smem + 16 * 1024 * 4);                // 16 x 1024 bf16
  float*  sRed = (float*)(smem + 16 * 1024 * 4 + 16 * 1024 * 2); // 4 waves x 32 x 8
  float*  sInv = sRed + 1024;                                    // 16

  const int idx = blockIdx.x;
  const int qt = idx & 63;
  const int h  = (idx >> 6) % HEADS;
  const int b  = idx / (64 * HEADS);
  const int t = threadIdx.x, w = t >> 5, lane = t & 31;
  const int col0 = lane & 15, roff = (lane >> 4) << 3;
  const float scale = 0.17677669529663687f;  // 1/sqrt(32)

  // ---- S = q @ k^T (each wave: 16 column tiles; constant trip count -> unrolled
  //      so the scheduler can overlap k-fragment loads with WMMA)
  const __bf16* qbase = q + ((size_t)(b * NSEQ + qt * 16)) * C_DIM + h * HD;
  const __bf16* kbase0 = k + ((size_t)(b * NSEQ + w * 256)) * C_DIM + h * HD;
  v16bf aq = frag_a(qbase, C_DIM);
#pragma unroll
  for (int i = 0; i < 16; ++i) {
    v16bf bk = frag_b(kbase0 + (size_t)(i * 16) * C_DIM, C_DIM);
    v8f accv;
#pragma unroll
    for (int e = 0; e < 8; ++e) accv[e] = 0.0f;
    accv = __builtin_amdgcn_wmma_f32_16x16x32_bf16(false, aq, false, bk,
                                                   (short)0, accv, false, false);
#pragma unroll
    for (int r = 0; r < 8; ++r)
      sS[(size_t)(r + roff) * NSEQ + (w * 16 + i) * 16 + col0] = accv[r] * scale;
  }
  __syncthreads();

  // ---- softmax: 8 threads per row
  {
    const int row = t >> 3, sub = t & 7;
    float* srow = sS + (size_t)row * NSEQ;
    float m = -3.4e38f;
    for (int c = sub; c < NSEQ; c += 8) m = fmaxf(m, srow[c]);
#pragma unroll
    for (int msk = 4; msk >= 1; msk >>= 1) m = fmaxf(m, __shfl_xor(m, msk));
    float ssum = 0.0f;
    __bf16* prow = sP + (size_t)row * NSEQ;
    for (int c = sub; c < NSEQ; c += 8) {
      float e = __expf(srow[c] - m);
      ssum += e;
      prow[c] = (__bf16)e;   // unnormalized; normalize in epilogue
    }
#pragma unroll
    for (int msk = 4; msk >= 1; msk >>= 1) ssum += __shfl_xor(ssum, msk);
    if (sub == 0) sInv[row] = 1.0f / ssum;
  }
  __syncthreads();

  // ---- O = P @ V : wave w -> col tile (w&1), K-half (w>>1); unrolled so LDS
  //      P-fragment loads and global V-fragment loads overlap the WMMA chain
  const int tn = w & 1, khalf = w >> 1;
  const __bf16* vbase = vT + ((size_t)(b * C_DIM) + h * HD + tn * 16) * NSEQ + khalf * 512;
  const __bf16* pbase = sP + khalf * 512;
  v8f acco;
#pragma unroll
  for (int e = 0; e < 8; ++e) acco[e] = 0.0f;
#pragma unroll
  for (int s = 0; s < 16; ++s) {
    v16bf ap = frag_a(pbase + s * 32, NSEQ);
    v16bf bv = frag_b(vbase + s * 32, NSEQ);
    acco = __builtin_amdgcn_wmma_f32_16x16x32_bf16(false, ap, false, bv,
                                                   (short)0, acco, false, false);
  }
#pragma unroll
  for (int e = 0; e < 8; ++e) sRed[(w * 32 + lane) * 8 + e] = acco[e];
  __syncthreads();
  if (w < 2) {
    __bf16* obase = o + ((size_t)(b * NSEQ + qt * 16)) * C_DIM + h * HD + tn * 16;
#pragma unroll
    for (int r = 0; r < 8; ++r) {
      const int row = r + roff;
      float val = (sRed[(w * 32 + lane) * 8 + r] + sRed[((w + 2) * 32 + lane) * 8 + r]) * sInv[row];
      obase[(size_t)row * C_DIM + col0] = (__bf16)val;
    }
  }
}

// ---------------------------------------------------------------- launch
extern "C" void kernel_launch(void* const* d_in, const int* in_sizes, int n_in,
                              void* d_out, int out_size, void* d_ws, size_t ws_size,
                              hipStream_t stream) {
  const float* x      = (const float*)d_in[0];
  const float* ln1_g  = (const float*)d_in[1];
  const float* ln1_b  = (const float*)d_in[2];
  const float* Wq     = (const float*)d_in[3];
  const float* bq     = (const float*)d_in[4];
  const float* off_k  = (const float*)d_in[5];
  const float* off_kb = (const float*)d_in[6];
  const float* offg   = (const float*)d_in[7];
  const float* offb   = (const float*)d_in[8];
  const float* Woff   = (const float*)d_in[9];
  const float* boff   = (const float*)d_in[10];
  const float* Wk     = (const float*)d_in[11];
  const float* bk     = (const float*)d_in[12];
  const float* Wv     = (const float*)d_in[13];
  const float* bv     = (const float*)d_in[14];
  const float* Wo     = (const float*)d_in[15];
  const float* bo     = (const float*)d_in[16];
  const float* ln2_g  = (const float*)d_in[17];
  const float* ln2_b  = (const float*)d_in[18];
  const float* W1     = (const float*)d_in[19];
  const float* b1     = (const float*)d_in[20];
  const float* W2     = (const float*)d_in[21];
  const float* b2     = (const float*)d_in[22];
  float* out = (float*)d_out;

  // bump allocator (all sizes are multiples of 256B)
  size_t off = 0;
  auto alloc = [&](size_t bytes) -> void* {
    off = (off + 255) & ~(size_t)255;
    void* p = (char*)d_ws + off;
    off += bytes;
    return p;
  };
  float*  xn_f = (float*)alloc((size_t)BN * C_DIM * 4);
  float*  q_f  = (float*)alloc((size_t)BN * C_DIM * 4);   // contiguous with xn_f
  float*  x2_f = (float*)alloc((size_t)BN * C_DIM * 4);
  __bf16* xn_b = (__bf16*)alloc((size_t)BN * C_DIM * 2);
  __bf16* q_b  = (__bf16*)alloc((size_t)BN * C_DIM * 2);
  __bf16* k_b  = (__bf16*)alloc((size_t)BN * C_DIM * 2);
  __bf16* vT_b = (__bf16*)alloc((size_t)BATCH * C_DIM * NSEQ * 2);
  __bf16* o_b  = (__bf16*)alloc((size_t)BN * C_DIM * 2);
  __bf16* Wq_b = (__bf16*)alloc((size_t)C_DIM * C_DIM * 2);
  __bf16* Wk_b = (__bf16*)alloc((size_t)C_DIM * C_DIM * 2);
  __bf16* Wv_b = (__bf16*)alloc((size_t)C_DIM * C_DIM * 2);
  __bf16* Wo_b = (__bf16*)alloc((size_t)C_DIM * C_DIM * 2);
  __bf16* W1_b = (__bf16*)alloc((size_t)MLP_DIM * C_DIM * 2);
  __bf16* W2_b = (__bf16*)alloc((size_t)C_DIM * MLP_DIM * 2);
  // aliases (lifetimes disjoint): xs reuses xn_b (xn_b dead after q GEMM);
  // y reuses q_b (dead after attention); h reuses xn_f+q_f (dead after offset kernel)
  __bf16* xs_b = xn_b;
  __bf16* y_b  = q_b;
  __bf16* h_b  = (__bf16*)xn_f;   // 50,331,648 B == xn_f + q_f exactly

  // 1) weights -> bf16
  {
    int n = C_DIM * C_DIM;
    cvt_bf16_kernel<<<(n + 255) / 256, 256, 0, stream>>>(Wq, Wq_b, n);
    cvt_bf16_kernel<<<(n + 255) / 256, 256, 0, stream>>>(Wk, Wk_b, n);
    cvt_bf16_kernel<<<(n + 255) / 256, 256, 0, stream>>>(Wv, Wv_b, n);
    cvt_bf16_kernel<<<(n + 255) / 256, 256, 0, stream>>>(Wo, Wo_b, n);
    int n2 = MLP_DIM * C_DIM;
    cvt_bf16_kernel<<<(n2 + 255) / 256, 256, 0, stream>>>(W1, W1_b, n2);
    cvt_bf16_kernel<<<(n2 + 255) / 256, 256, 0, stream>>>(W2, W2_b, n2);
  }

  auto gemm_blocks = [](int M, int N) { return ((M / 32) * (N / 64) + 7) / 8; };

  // 2) LN1
  ln_kernel<<<BN, 128, 0, stream>>>(x, ln1_g, ln1_b, xn_f, xn_b);
  // 3) q = xn @ Wq^T + bq  (f32 + bf16)
  gemm_bf16_kernel<1><<<gemm_blocks(BN, C_DIM), 256, 0, stream>>>(
      xn_b, Wq_b, bq, q_f, q_b, nullptr, BN, C_DIM, C_DIM);
  // 4) offset network + grid sample -> xs (bf16)
  offset_sample_kernel<<<BN, 128, 0, stream>>>(q_f, off_k, off_kb, offg, offb,
                                               Woff, boff, xn_f, xs_b);
  // 5) k = xs @ Wk^T + bk (bf16); v stored transposed
  gemm_bf16_kernel<2><<<gemm_blocks(BN, C_DIM), 256, 0, stream>>>(
      xs_b, Wk_b, bk, nullptr, k_b, nullptr, BN, C_DIM, C_DIM);
  gemm_bf16_kernel<3><<<gemm_blocks(BN, C_DIM), 256, 0, stream>>>(
      xs_b, Wv_b, bv, nullptr, vT_b, nullptr, BN, C_DIM, C_DIM);
  // 6) attention
  attn_kernel<<<BATCH * HEADS * 64, 128, ATTN_SMEM, stream>>>(q_b, k_b, vT_b, o_b);
  // 7) x2 = x + o @ Wo^T + bo
  gemm_bf16_kernel<5><<<gemm_blocks(BN, C_DIM), 256, 0, stream>>>(
      o_b, Wo_b, bo, x2_f, nullptr, x, BN, C_DIM, C_DIM);
  // 8) LN2 -> y (bf16)
  ln_kernel<<<BN, 128, 0, stream>>>(x2_f, ln2_g, ln2_b, nullptr, y_b);
  // 9) h = gelu(y @ W1^T + b1) (bf16)
  gemm_bf16_kernel<4><<<gemm_blocks(BN, MLP_DIM), 256, 0, stream>>>(
      y_b, W1_b, b1, nullptr, h_b, nullptr, BN, MLP_DIM, C_DIM);
  // 10) out = x2 + h @ W2^T + b2
  gemm_bf16_kernel<5><<<gemm_blocks(BN, C_DIM), 256, 0, stream>>>(
      h_b, W2_b, b2, out, nullptr, x2_f, BN, C_DIM, MLP_DIM);

  (void)in_sizes; (void)n_in; (void)out_size; (void)ws_size;
}
